// RSSM_180388627102
// MI455X (gfx1250) — matrix operand order repaired and checked by reading, not compiled, not run on hardware
//
#include <hip/hip_runtime.h>
#include <hip/hip_bf16.h>

typedef unsigned short u16;
typedef __bf16 bf16x16 __attribute__((ext_vector_type(16)));
typedef float  floatx8 __attribute__((ext_vector_type(8)));

#define MROWS 6144      // S*B*N
#define NBATCH 1024     // S*B
#define NSLOT 6

// ---------- helpers ----------
__device__ __forceinline__ u16 f2bf(float f) {
  unsigned int u = __float_as_uint(f);
  u += 0x7FFFu + ((u >> 16) & 1u);          // round-to-nearest-even
  return (u16)(u >> 16);
}

// ---------- weight convert: f32 [K,N] -> bf16 transposed [N,Kpad] ----------
// LDS-tiled 32x32 transpose; rows k >= Ksrc are zero padding.
__global__ __launch_bounds__(256) void cast_transpose(
    const float* __restrict__ src, u16* __restrict__ dst,
    int Ksrc, int Kpad, int N) {
  __shared__ u16 tile[32][33];
  const int kb = blockIdx.x * 32, nb = blockIdx.y * 32;
  const int tx = threadIdx.x & 31, ty = threadIdx.x >> 5;   // ty = 0..7
  #pragma unroll
  for (int r = 0; r < 4; ++r) {
    const int k = kb + ty + 8 * r;
    float v = (k < Ksrc) ? src[(size_t)k * N + nb + tx] : 0.0f;
    tile[ty + 8 * r][tx] = f2bf(v);
  }
  __syncthreads();
  #pragma unroll
  for (int r = 0; r < 4; ++r) {
    const int n = nb + ty + 8 * r;
    dst[(size_t)n * Kpad + kb + tx] = tile[tx][ty + 8 * r];
  }
}

// ---------- strided f32 -> bf16 cast (for building concat inputs) ----------
__global__ __launch_bounds__(256) void cast_strided(
    const float* __restrict__ src, int lds, u16* __restrict__ dst, int ldd,
    int total, int cols) {
  int i = blockIdx.x * 256 + threadIdx.x;
  if (i >= total) return;
  int row = i / cols, col = i - row * cols;
  dst[(size_t)row * ldd + col] = f2bf(src[(size_t)row * lds + col]);
}

// ---------- build [stoch, action_broadcast, zero-pad] (6144 x 1056) ----------
__global__ __launch_bounds__(256) void build_xcat(
    const float* __restrict__ stoch, const float* __restrict__ action,
    u16* __restrict__ dst) {
  int i = blockIdx.x * 256 + threadIdx.x;
  const int total = MROWS * 1056;
  if (i >= total) return;
  int row = i / 1056, col = i - row * 1056;
  float v = 0.0f;
  if (col < 1024) v = stoch[(size_t)row * 1024 + col];
  else if (col < 1030) { int b = row / NSLOT; v = action[b * 6 + (col - 1024)]; }
  dst[i] = f2bf(v);
}

// ---------- bf16 WMMA GEMM: C[M,N] = A[M,K] @ Bt[N,K]^T + bias + residual ----
// Bt is K-major (pre-transposed) weight. Block = 128 threads (4 waves in 2x2),
// block tile 64x128, wave tile 32x64, K step 32. Straight-line b128 staging.
__global__ __launch_bounds__(128) void gemm_bf16(
    const u16* __restrict__ A, int lda,
    const u16* __restrict__ Bt,          // [N][K] bf16, row stride = K
    const float* __restrict__ bias,
    const float* __restrict__ residual, int ldr,
    float* __restrict__ C, int ldc, int K) {
  __shared__ __align__(16) u16 As[64 * 32];    // 64 m-rows x 32 k
  __shared__ __align__(16) u16 Bs[128 * 32];   // 128 n-cols x 32 k
  const int tid  = threadIdx.x;
  const int wave = tid >> 5;
  const int lane = tid & 31;
  const int half = lane >> 4;
  const int l15  = lane & 15;
  const int wm   = wave >> 1;        // 0..1 : 32-row group
  const int wn   = wave & 1;         // 0..1 : 64-col group
  const int m0 = blockIdx.y * 64;
  const int n0 = blockIdx.x * 128;

  // staging chunk coordinates (16B chunks, straight-line, no bound checks)
  const int srow = tid >> 2;         // 0..31
  const int koff = (tid & 3) * 8;    // 0,8,16,24

  floatx8 acc[2][4];
  #pragma unroll
  for (int at = 0; at < 2; ++at)
    #pragma unroll
    for (int nt = 0; nt < 4; ++nt) {
      floatx8 z = {0.f, 0.f, 0.f, 0.f, 0.f, 0.f, 0.f, 0.f};
      acc[at][nt] = z;
    }

  const u16* ap0 = A  + (size_t)(m0 + srow)       * lda + koff;
  const u16* ap1 = A  + (size_t)(m0 + srow + 32)  * lda + koff;
  const u16* bp0 = Bt + (size_t)(n0 + srow)       * K   + koff;
  const u16* bp1 = Bt + (size_t)(n0 + srow + 32)  * K   + koff;
  const u16* bp2 = Bt + (size_t)(n0 + srow + 64)  * K   + koff;
  const u16* bp3 = Bt + (size_t)(n0 + srow + 96)  * K   + koff;

  for (int k0 = 0; k0 < K; k0 += 32) {
    // ---- stage: issue all global loads, then all LDS stores ----
    uint4 a0 = *(const uint4*)(ap0 + k0);
    uint4 a1 = *(const uint4*)(ap1 + k0);
    uint4 b0 = *(const uint4*)(bp0 + k0);
    uint4 b1 = *(const uint4*)(bp1 + k0);
    uint4 b2 = *(const uint4*)(bp2 + k0);
    uint4 b3 = *(const uint4*)(bp3 + k0);
    *(uint4*)&As[(srow)      * 32 + koff] = a0;
    *(uint4*)&As[(srow + 32) * 32 + koff] = a1;
    *(uint4*)&Bs[(srow)      * 32 + koff] = b0;
    *(uint4*)&Bs[(srow + 32) * 32 + koff] = b1;
    *(uint4*)&Bs[(srow + 64) * 32 + koff] = b2;
    *(uint4*)&Bs[(srow + 96) * 32 + koff] = b3;
    __syncthreads();

    // ---- load all fragments, then burst 8 WMMAs ----
    union { uint4 q[2]; bf16x16 v; } af[2];
    #pragma unroll
    for (int at = 0; at < 2; ++at) {
      const int arow = wm * 32 + at * 16 + l15;
      af[at].q[0] = *(const uint4*)&As[arow * 32 + half * 8];
      af[at].q[1] = *(const uint4*)&As[arow * 32 + 16 + half * 8];
    }
    union { uint4 q[2]; bf16x16 v; } bfr[4];
    #pragma unroll
    for (int nt = 0; nt < 4; ++nt) {
      const int ncol = wn * 64 + nt * 16 + l15;
      bfr[nt].q[0] = *(const uint4*)&Bs[ncol * 32 + half * 8];
      bfr[nt].q[1] = *(const uint4*)&Bs[ncol * 32 + 16 + half * 8];
    }
    #pragma unroll
    for (int at = 0; at < 2; ++at)
      #pragma unroll
      for (int nt = 0; nt < 4; ++nt)
        acc[at][nt] = __builtin_amdgcn_wmma_f32_16x16x32_bf16(
            false, af[at].v, false, bfr[nt].v, (short)0, acc[at][nt], false, false);
    __syncthreads();
  }

  // ---- epilogue: VGPR r -> M = half*8 + r ; N = lane&15 ----
  #pragma unroll
  for (int at = 0; at < 2; ++at)
    #pragma unroll
    for (int nt = 0; nt < 4; ++nt) {
      floatx8 a = acc[at][nt];
      const int col = n0 + wn * 64 + nt * 16 + l15;
      const float bv = bias ? bias[col] : 0.0f;
      #pragma unroll
      for (int r = 0; r < 8; ++r) {
        const int row = m0 + wm * 32 + at * 16 + half * 8 + r;
        float val = a[r] + bv;
        if (residual) val += residual[(size_t)row * ldr + col];
        C[(size_t)row * ldc + col] = val;
      }
    }
}

// ---------- LayerNorm (+optional ELU) -> strided bf16 ----------
__global__ __launch_bounds__(256) void ln_bf16(
    const float* __restrict__ x, int W,
    const float* __restrict__ g, const float* __restrict__ bvec,
    u16* __restrict__ out, int ldo, int act) {
  __shared__ float s1[256], s2[256];
  const int row = blockIdx.x, tid = threadIdx.x;
  const float* xr = x + (size_t)row * W;
  float a = 0.f, b = 0.f;
  for (int j = tid; j < W; j += 256) { float v = xr[j]; a += v; b += v * v; }
  s1[tid] = a; s2[tid] = b; __syncthreads();
  for (int o = 128; o; o >>= 1) {
    if (tid < o) { s1[tid] += s1[tid + o]; s2[tid] += s2[tid + o]; }
    __syncthreads();
  }
  const float mean = s1[0] / (float)W;
  const float var  = s2[0] / (float)W - mean * mean;
  const float rstd = rsqrtf(var + 1e-5f);
  for (int j = tid; j < W; j += 256) {
    float y = (xr[j] - mean) * rstd * g[j] + bvec[j];
    if (act) y = (y > 0.f) ? y : expm1f(y);   // ELU
    out[(size_t)row * ldo + j] = f2bf(y);
  }
}

// ---------- fused GRU: LN(3072) -> gates -> h ----------
__global__ __launch_bounds__(256) void gru_kernel(
    const float* __restrict__ gmat, const float* __restrict__ lng,
    const float* __restrict__ lnb, const float* __restrict__ hprev,
    float* __restrict__ outPrior, float* __restrict__ dpost) {
  __shared__ float s1[256], s2[256];
  const int row = blockIdx.x, tid = threadIdx.x;
  const float* g = gmat + (size_t)row * 3072;
  float a = 0.f, b = 0.f;
  for (int j = tid; j < 3072; j += 256) { float v = g[j]; a += v; b += v * v; }
  s1[tid] = a; s2[tid] = b; __syncthreads();
  for (int o = 128; o; o >>= 1) {
    if (tid < o) { s1[tid] += s1[tid + o]; s2[tid] += s2[tid + o]; }
    __syncthreads();
  }
  const float mean = s1[0] / 3072.f;
  const float var  = s2[0] / 3072.f - mean * mean;
  const float rstd = rsqrtf(var + 1e-5f);
  for (int j = tid; j < 1024; j += 256) {
    float pr = (g[j]         - mean) * rstd * lng[j]         + lnb[j];
    float pc = (g[1024 + j]  - mean) * rstd * lng[1024 + j]  + lnb[1024 + j];
    float pu = (g[2048 + j]  - mean) * rstd * lng[2048 + j]  + lnb[2048 + j];
    float r  = 1.f / (1.f + __expf(-pr));
    float c  = tanhf(r * pc);
    float uu = 1.f / (1.f + __expf(-(pu - 1.f)));
    float h  = uu * c + (1.f - uu) * hprev[(size_t)row * 1024 + j];
    outPrior[(size_t)row * 1024 + j] = h;
    dpost[(size_t)row * 1024 + j]    = h;
  }
}

// ---------- 6-slot attention: softmax(scale*qk+eps) mixed with identity ----------
__global__ __launch_bounds__(256) void attn6(
    const float* __restrict__ q, int ldq, const float* __restrict__ k, int ldk,
    const float* __restrict__ v, int ldv, float* __restrict__ out, int ldo,
    int hd, float scale, float coeff) {
  __shared__ float logits[36];
  __shared__ float attnv[36];
  const int b = blockIdx.x;
  const int base = b * NSLOT;
  const int tid = threadIdx.x, wave = tid >> 5, lane = tid & 31;

  for (int p = wave; p < 36; p += 8) {
    const int i = p / 6, j = p - 6 * (p / 6);
    const float* qr = q + (size_t)(base + i) * ldq;
    const float* kr = k + (size_t)(base + j) * ldk;
    float s = 0.f;
    for (int h = lane; h < hd; h += 32) s += qr[h] * kr[h];
    #pragma unroll
    for (int off = 16; off; off >>= 1) s += __shfl_down(s, off, 32);
    if (lane == 0) logits[p] = s;
  }
  __syncthreads();
  if (tid < 6) {
    const int i = tid;
    float l[6], mx = -1e30f;
    #pragma unroll
    for (int j = 0; j < 6; ++j) {
      l[j] = scale * logits[i * 6 + j] + 1e-8f;
      mx = fmaxf(mx, l[j]);
    }
    float sum = 0.f;
    #pragma unroll
    for (int j = 0; j < 6; ++j) { l[j] = __expf(l[j] - mx); sum += l[j]; }
    const float inv = 1.f / sum;   // softmax; reference's renorm is /1 (identity)
    #pragma unroll
    for (int j = 0; j < 6; ++j) {
      float av = l[j] * inv;
      attnv[i * 6 + j] = coeff * av + (1.f - coeff) * ((i == j) ? 1.f : 0.f);
    }
  }
  __syncthreads();
  for (int idx = tid; idx < NSLOT * hd; idx += 256) {
    const int i = idx / hd, h = idx - i * hd;
    float s = 0.f;
    #pragma unroll
    for (int j = 0; j < 6; ++j)
      s += attnv[i * 6 + j] * v[(size_t)(base + j) * ldv + h];
    out[(size_t)(base + i) * ldo + h] = s;
  }
}

// =====================================================================
extern "C" void kernel_launch(void* const* d_in, const int* in_sizes, int n_in,
                              void* d_out, int out_size, void* d_ws, size_t ws_size,
                              hipStream_t stream) {
  (void)in_sizes; (void)n_in; (void)out_size; (void)ws_size;
  const int M = MROWS;

  // ---- inputs ----
  const float* determ = (const float*)d_in[0];
  const float* stoch  = (const float*)d_in[1];
  const float* embed  = (const float*)d_in[3];
  const float* action = (const float*)d_in[4];
  const float* W1   = (const float*)d_in[5];
  const float* b1   = (const float*)d_in[6];
  const float* ln1g = (const float*)d_in[7];
  const float* ln1b = (const float*)d_in[8];
  const float* Wg   = (const float*)d_in[9];
  const float* lngg = (const float*)d_in[10];
  const float* lngb = (const float*)d_in[11];
  const float* Wqkv = (const float*)d_in[12];
  const float* bqkv = (const float*)d_in[13];
  const float* png  = (const float*)d_in[14];
  const float* pnb  = (const float*)d_in[15];
  const float* Wfc  = (const float*)d_in[16];
  const float* bfc  = (const float*)d_in[17];
  const float* fng  = (const float*)d_in[18];
  const float* fnb  = (const float*)d_in[19];
  const float* Wp1  = (const float*)d_in[20];
  const float* bp1  = (const float*)d_in[21];
  const float* lnpg = (const float*)d_in[22];
  const float* lnpb = (const float*)d_in[23];
  const float* Wp2  = (const float*)d_in[24];
  const float* bp2  = (const float*)d_in[25];
  const float* Wos  = (const float*)d_in[26];
  const float* bos  = (const float*)d_in[27];
  const float* Wo   = (const float*)d_in[28];
  const float* bo   = (const float*)d_in[29];
  const float* pnog = (const float*)d_in[30];
  const float* pnob = (const float*)d_in[31];
  const float* Ws1  = (const float*)d_in[32];
  const float* bs1  = (const float*)d_in[33];
  const float* lnsg = (const float*)d_in[34];
  const float* lnsb = (const float*)d_in[35];
  const float* Ws2  = (const float*)d_in[36];
  const float* bs2  = (const float*)d_in[37];

  float* outDeterm = (float*)d_out;
  float* outPrior  = (float*)d_out + (size_t)M * 1024;
  float* outPost   = (float*)d_out + (size_t)2 * M * 1024;

  // ---- workspace bump allocator ----
  char* base = (char*)d_ws;
  size_t off = 0;
  auto alloc = [&](size_t bytes) -> void* {
    void* p = base + off;
    off = (off + bytes + 255) & ~(size_t)255;
    return p;
  };
  // transposed bf16 weights: [N][Kpad]
  u16* W1b  = (u16*)alloc((size_t)1024 * 1056 * 2);
  u16* Wgb  = (u16*)alloc((size_t)3072 * 2048 * 2);
  u16* Wqkb = (u16*)alloc((size_t)3072 * 1024 * 2);
  u16* Wfcb = (u16*)alloc((size_t)1024 * 1024 * 2);
  u16* Wp1b = (u16*)alloc((size_t)1024 * 1024 * 2);
  u16* Wp2b = (u16*)alloc((size_t)1024 * 1024 * 2);
  u16* Wosb = (u16*)alloc((size_t)1536 * 1024 * 2);
  u16* Wob  = (u16*)alloc((size_t)1536 * 1536 * 2);
  u16* Ws1b = (u16*)alloc((size_t)1024 * 2560 * 2);
  u16* Ws2b = (u16*)alloc((size_t)1024 * 1024 * 2);
  float* F0 = (float*)alloc((size_t)M * 3072 * 4);   // gates / qkv / emb2 (aliased)
  float* F1 = (float*)alloc((size_t)M * 1536 * 4);   // upd / k2 / z1
  float* F2 = (float*)alloc((size_t)M * 1536 * 4);   // t0 / y / q2
  float* DP = (float*)alloc((size_t)M * 1024 * 4);   // determ_post
  u16*  B0  = (u16*)alloc((size_t)M * 2560 * 2);     // concat buffers (bf16)
  u16*  B1  = (u16*)alloc((size_t)M * 1536 * 2);     // LN outputs (bf16)
  float* F3 = F0;                                    // emb2 aliases F0 (free then)

  const dim3 blk(256);
  auto g1 = [](int n) { return dim3((n + 255) / 256); };
  auto wtrans = [&](const float* src, u16* dst, int Ksrc, int Kpad, int N) {
    dim3 grid(Kpad / 32, N / 32);
    cast_transpose<<<grid, blk, 0, stream>>>(src, dst, Ksrc, Kpad, N);
  };
  auto gemm = [&](const u16* A, int lda, const u16* Bt,
                  const float* bias, const float* res, int ldr,
                  float* C, int ldc, int N, int K) {
    dim3 grid(N / 128, M / 64);
    gemm_bf16<<<grid, dim3(128), 0, stream>>>(A, lda, Bt, bias, res, ldr, C, ldc, K);
  };

  const float SCALE = 0.03125f;   // 1024^-0.5
  // ---- convert weights: f32 [K,N] -> bf16 [N,Kpad] (K padded to mult of 32) ----
  wtrans(W1,  W1b,  1030, 1056, 1024);
  wtrans(Wg,  Wgb,  2048, 2048, 3072);
  wtrans(Wqkv,Wqkb, 1024, 1024, 3072);
  wtrans(Wfc, Wfcb, 1024, 1024, 1024);
  wtrans(Wp1, Wp1b, 1024, 1024, 1024);
  wtrans(Wp2, Wp2b, 1024, 1024, 1024);
  wtrans(Wos, Wosb, 1024, 1024, 1536);
  wtrans(Wo,  Wob,  1536, 1536, 1536);
  wtrans(Ws1, Ws1b, 2560, 2560, 1024);
  wtrans(Ws2, Ws2b, 1024, 1024, 1024);

  // ---- predict_next ----
  build_xcat<<<g1(M*1056), blk, 0, stream>>>(stoch, action, B0);
  gemm(B0, 1056, W1b, b1, nullptr, 0, F2, 1024, 1024, 1056);
  ln_bf16<<<M, blk, 0, stream>>>(F2, 1024, ln1g, ln1b, B0, 2048, 1);       // x -> cols [0,1024)
  cast_strided<<<g1(M*1024), blk, 0, stream>>>(determ, 1024, B0 + 1024, 2048, M*1024, 1024);
  gemm(B0, 2048, Wgb, nullptr, nullptr, 0, F0, 3072, 3072, 2048);
  gru_kernel<<<M, blk, 0, stream>>>(F0, lngg, lngb, determ, outDeterm, DP);

  // ---- 3 attention blocks ----
  for (int it = 0; it < 3; ++it) {
    ln_bf16<<<M, blk, 0, stream>>>(DP, 1024, png, pnb, B1, 1024, 0);
    gemm(B1, 1024, Wqkb, bqkv, nullptr, 0, F0, 3072, 3072, 1024);
    attn6<<<NBATCH, blk, 0, stream>>>(F0, 3072, F0 + 1024, 3072, F0 + 2048, 3072,
                                      F1, 1024, 1024, SCALE, 0.5f);
    ln_bf16<<<M, blk, 0, stream>>>(F1, 1024, fng, fnb, B1, 1024, 0);
    gemm(B1, 1024, Wfcb, bfc, DP, 1024, DP, 1024, 1024, 1024);             // residual add
  }

  // ---- prior head ----
  cast_strided<<<g1(M*1024), blk, 0, stream>>>(DP, 1024, B1, 1024, M*1024, 1024);
  gemm(B1, 1024, Wp1b, bp1, nullptr, 0, F2, 1024, 1024, 1024);
  ln_bf16<<<M, blk, 0, stream>>>(F2, 1024, lnpg, lnpb, B1, 1024, 1);
  gemm(B1, 1024, Wp2b, bp2, nullptr, 0, outPrior, 1024, 1024, 1024);

  // ---- update_current ----
  ln_bf16<<<M, blk, 0, stream>>>(DP, 1024, png, pnb, B1, 1024, 0);
  gemm(B1, 1024, Wosb, bos, nullptr, 0, F1, 1536, 1536, 1024);             // k2
  ln_bf16<<<M, blk, 0, stream>>>(embed, 1536, pnog, pnob, B1, 1536, 0);
  gemm(B1, 1536, Wob, bo, nullptr, 0, F2, 1536, 1536, 1536);               // q2
  attn6<<<NBATCH, blk, 0, stream>>>(F2, 1536, F1, 1536, embed, 1536,
                                    F3, 1536, 1536, SCALE, 1.0f);          // c2 = 1.0
  cast_strided<<<g1(M*1024), blk, 0, stream>>>(DP, 1024, B0,        2560, M*1024, 1024);
  cast_strided<<<g1(M*1536), blk, 0, stream>>>(F3, 1536, B0 + 1024, 2560, M*1536, 1536);
  gemm(B0, 2560, Ws1b, bs1, nullptr, 0, F1, 1024, 1024, 2560);
  ln_bf16<<<M, blk, 0, stream>>>(F1, 1024, lnsg, lnsb, B1, 1024, 1);
  gemm(B1, 1024, Ws2b, bs2, nullptr, 0, outPost, 1024, 1024, 1024);
}
